// TracablePrefillV9_66709432041910
// MI455X (gfx1250) — matrix operand order repaired and checked by reading, not compiled, not run on hardware
//
#include <hip/hip_runtime.h>

// ---------------- model constants ----------------
#define B_    8
#define S_    139          // MAXT + 11
#define H_    1024
#define NH_   16
#define NKV_  4
#define HD_   64
#define FF_   4096
#define L_    12
#define CV_   8448
#define TD_   2048
#define MAXT_ 128
#define NTOK_ (B_ * S_)          // 1112
#define NEMB_ (B_ * (3 + MAXT_)) // 1048 gathered embedding rows
#define NTOKP_ 1152              // NTOK padded to 64-row M-tiles
#define NEMBP_ 1088              // NEMB padded to 64-row M-tiles

// WMMA vector types (gfx1250, wave32)
typedef __attribute__((ext_vector_type(16))) __bf16 bf16x16;
typedef __attribute__((ext_vector_type(8)))  float  f32x8;

__device__ __forceinline__ f32x8 wmma_bf16(bf16x16 a, bf16x16 b, f32x8 c) {
  // D = A(16x32) * B(32x16) + C(16x16), f32 accumulate
  return __builtin_amdgcn_wmma_f32_16x16x32_bf16(false, a, false, b, (short)0, c,
                                                 false, false);
}

// pack 4 f32 -> 4 bf16, one 8-byte LDS store
__device__ __forceinline__ void st_bf16x4(__bf16* dst, float4 v) {
  union { __bf16 t[4]; uint2 u; } x;
  x.t[0] = (__bf16)v.x; x.t[1] = (__bf16)v.y;
  x.t[2] = (__bf16)v.z; x.t[3] = (__bf16)v.w;
  *(uint2*)dst = x.u;
}
// 16 contiguous bf16 (two ds_load_b128)
__device__ __forceinline__ bf16x16 ld_b16x16(const __bf16* p) {
  union { uint4 u[2]; bf16x16 v; } x;
  x.u[0] = *(const uint4*)p;
  x.u[1] = *(const uint4*)(p + 8);
  return x.v;
}
// two 8-element bf16 chunks (A-fragment pattern)
__device__ __forceinline__ bf16x16 ld_b8x2(const __bf16* p0, const __bf16* p1) {
  union { uint4 u[2]; bf16x16 v; } x;
  x.u[0] = *(const uint4*)p0;
  x.u[1] = *(const uint4*)p1;
  return x.v;
}
// build bf16 fragment from two 8-float global chunks
__device__ __forceinline__ bf16x16 cvt_frag(const float* p0, const float* p1) {
  float4 a0 = *(const float4*)p0, a1 = *(const float4*)(p0 + 4);
  float4 b0 = *(const float4*)p1, b1 = *(const float4*)(p1 + 4);
  bf16x16 r;
  r[0] = (__bf16)a0.x; r[1] = (__bf16)a0.y; r[2] = (__bf16)a0.z; r[3] = (__bf16)a0.w;
  r[4] = (__bf16)a1.x; r[5] = (__bf16)a1.y; r[6] = (__bf16)a1.z; r[7] = (__bf16)a1.w;
  r[8] = (__bf16)b0.x; r[9] = (__bf16)b0.y; r[10] = (__bf16)b0.z; r[11] = (__bf16)b0.w;
  r[12] = (__bf16)b1.x; r[13] = (__bf16)b1.y; r[14] = (__bf16)b1.z; r[15] = (__bf16)b1.w;
  return r;
}

// ============================================================
// 1) gather embedding rows: out[m][:] = text_emb[id(m)][:]
// ============================================================
__global__ __launch_bounds__(256) void k_gather_rows(
    const float* __restrict__ emb, const int* __restrict__ role_ids,
    const int* __restrict__ text_ids, float* __restrict__ out) {
  int m = blockIdx.x;
  int b = m / 131, r = m % 131;
  int id = (r < 3) ? role_ids[b * 3 + r] : text_ids[b * MAXT_ + (r - 3)];
  const float* src = emb + (size_t)id * TD_;
  float* dst = out + (size_t)m * TD_;
  for (int e = threadIdx.x * 4; e < TD_; e += 1024)
    *(float4*)(dst + e) = *(const float4*)(src + e);
}

// ============================================================
// 2) generic bf16-WMMA GEMM: C[M,N] = A[M,K] @ W[K,N] (+ R)
//    128 threads (4 waves), tile 64(M) x 64(N) x 32(K)
//    A rows must be readable up to gridDim.y*64 (padded workspace)
// ============================================================
__global__ __launch_bounds__(128) void k_gemm(
    const float* __restrict__ A, const float* __restrict__ W,
    const float* __restrict__ R, float* __restrict__ C, int M, int N, int K) {
  __shared__ __align__(16) __bf16 sA[64 * 32];  // [m][k]
  __shared__ __align__(16) __bf16 sB[64 * 32];  // [n][k] (transposed)
  const int tid = threadIdx.x;
  const int wave = tid >> 5, lane = tid & 31;
  const int hi = lane >> 4, ln = lane & 15;
  const int m0 = blockIdx.y * 64;
  const int nb = blockIdx.x * 64;
  const int arow = tid >> 3, acol = (tid & 7) * 4;  // A slot (row+16*it, col)
  const int bk = tid >> 4, bn = (tid & 15) * 4;     // W slot (k+8*it, 4 cols)
  float4 ra[4], rb[4];
#pragma unroll
  for (int it = 0; it < 4; ++it) {  // prefetch k-tile 0
    ra[it] = *(const float4*)(A + (size_t)(m0 + arow + it * 16) * K + acol);
    rb[it] = *(const float4*)(W + (size_t)(bk + it * 8) * N + nb + bn);
  }
  f32x8 acc[4] = {};
  for (int k0 = 0; k0 < K; k0 += 32) {
#pragma unroll
    for (int it = 0; it < 4; ++it) {  // regs -> LDS (bf16)
      st_bf16x4(&sA[(arow + it * 16) * 32 + acol], ra[it]);
      int kk = bk + it * 8;
      sB[(bn + 0) * 32 + kk] = (__bf16)rb[it].x;
      sB[(bn + 1) * 32 + kk] = (__bf16)rb[it].y;
      sB[(bn + 2) * 32 + kk] = (__bf16)rb[it].z;
      sB[(bn + 3) * 32 + kk] = (__bf16)rb[it].w;
    }
    __syncthreads();
    if (k0 + 32 < K) {  // prefetch next k-tile while WMMAs run
#pragma unroll
      for (int it = 0; it < 4; ++it) {
        ra[it] = *(const float4*)(A + (size_t)(m0 + arow + it * 16) * K + (k0 + 32) + acol);
        rb[it] = *(const float4*)(W + (size_t)(k0 + 32 + bk + it * 8) * N + nb + bn);
      }
    }
    bf16x16 bf = ld_b16x16(&sB[(wave * 16 + ln) * 32 + 16 * hi]);
#pragma unroll
    for (int rt = 0; rt < 4; ++rt) {
      const __bf16* ap = &sA[(rt * 16 + ln) * 32];
      bf16x16 a = ld_b16x16(ap + 16 * hi);
      acc[rt] = wmma_bf16(a, bf, acc[rt]);
    }
    __syncthreads();
  }
  const int gn = nb + wave * 16 + ln;
#pragma unroll
  for (int rt = 0; rt < 4; ++rt) {
#pragma unroll
    for (int j = 0; j < 8; ++j) {  // C lane: row = rt*16 + j + 8*hi, col = ln
      int gm = m0 + rt * 16 + 8 * hi + j;
      if (gm < M) {
        size_t idx = (size_t)gm * N + gn;
        float v = acc[rt][j];
        if (R) v += R[idx];
        C[idx] = v;
      }
    }
  }
}

// ============================================================
// 3) fused gate/up GEMM: C = silu(A@Wg) * (A@Wu)
// ============================================================
__global__ __launch_bounds__(128) void k_gemm_gateup(
    const float* __restrict__ A, const float* __restrict__ Wg,
    const float* __restrict__ Wu, float* __restrict__ C, int M, int N, int K) {
  __shared__ __align__(16) __bf16 sA[64 * 32];
  __shared__ __align__(16) __bf16 sBg[64 * 32];
  __shared__ __align__(16) __bf16 sBu[64 * 32];
  const int tid = threadIdx.x;
  const int wave = tid >> 5, lane = tid & 31;
  const int hi = lane >> 4, ln = lane & 15;
  const int m0 = blockIdx.y * 64;
  const int nb = blockIdx.x * 64;
  const int arow = tid >> 3, acol = (tid & 7) * 4;
  const int bk = tid >> 4, bn = (tid & 15) * 4;
  float4 ra[4], rg[4], ru[4];
#pragma unroll
  for (int it = 0; it < 4; ++it) {
    ra[it] = *(const float4*)(A + (size_t)(m0 + arow + it * 16) * K + acol);
    rg[it] = *(const float4*)(Wg + (size_t)(bk + it * 8) * N + nb + bn);
    ru[it] = *(const float4*)(Wu + (size_t)(bk + it * 8) * N + nb + bn);
  }
  f32x8 accg[4] = {}, accu[4] = {};
  for (int k0 = 0; k0 < K; k0 += 32) {
#pragma unroll
    for (int it = 0; it < 4; ++it) {
      st_bf16x4(&sA[(arow + it * 16) * 32 + acol], ra[it]);
      int kk = bk + it * 8;
      sBg[(bn + 0) * 32 + kk] = (__bf16)rg[it].x;
      sBg[(bn + 1) * 32 + kk] = (__bf16)rg[it].y;
      sBg[(bn + 2) * 32 + kk] = (__bf16)rg[it].z;
      sBg[(bn + 3) * 32 + kk] = (__bf16)rg[it].w;
      sBu[(bn + 0) * 32 + kk] = (__bf16)ru[it].x;
      sBu[(bn + 1) * 32 + kk] = (__bf16)ru[it].y;
      sBu[(bn + 2) * 32 + kk] = (__bf16)ru[it].z;
      sBu[(bn + 3) * 32 + kk] = (__bf16)ru[it].w;
    }
    __syncthreads();
    if (k0 + 32 < K) {
#pragma unroll
      for (int it = 0; it < 4; ++it) {
        ra[it] = *(const float4*)(A + (size_t)(m0 + arow + it * 16) * K + (k0 + 32) + acol);
        rg[it] = *(const float4*)(Wg + (size_t)(k0 + 32 + bk + it * 8) * N + nb + bn);
        ru[it] = *(const float4*)(Wu + (size_t)(k0 + 32 + bk + it * 8) * N + nb + bn);
      }
    }
    bf16x16 bg = ld_b16x16(&sBg[(wave * 16 + ln) * 32 + 16 * hi]);
    bf16x16 bu = ld_b16x16(&sBu[(wave * 16 + ln) * 32 + 16 * hi]);
#pragma unroll
    for (int rt = 0; rt < 4; ++rt) {
      const __bf16* ap = &sA[(rt * 16 + ln) * 32];
      bf16x16 a = ld_b16x16(ap + 16 * hi);
      accg[rt] = wmma_bf16(a, bg, accg[rt]);
      accu[rt] = wmma_bf16(a, bu, accu[rt]);
    }
    __syncthreads();
  }
  const int gn = nb + wave * 16 + ln;
#pragma unroll
  for (int rt = 0; rt < 4; ++rt) {
#pragma unroll
    for (int j = 0; j < 8; ++j) {
      int gm = m0 + rt * 16 + 8 * hi + j;
      if (gm < M) {
        float g = accg[rt][j], u = accu[rt][j];
        C[(size_t)gm * N + gn] = (g / (1.f + __expf(-g))) * u;
      }
    }
  }
}

// ============================================================
// 4) sequence assembly into h (B,S,H), float4 per thread
// ============================================================
__global__ __launch_bounds__(256) void k_assemble(
    const float* __restrict__ embo, const float* __restrict__ codec,
    const float* __restrict__ bos, const float* __restrict__ pad,
    const float* __restrict__ eos, const float* __restrict__ spk,
    const int* __restrict__ tl, float* __restrict__ h) {
  int s = blockIdx.x, b = blockIdx.y;
  int len = tl[b];
  int e = threadIdx.x * 4;
  float4 v;
  const float4 padc = *(const float4*)(codec + (size_t)8196 * H_ + e);   // PADID
  const float4 bosc = *(const float4*)(codec + (size_t)8197 * H_ + e);   // BOSID
  const float4 padv = *(const float4*)(pad + b * H_ + e);
  if (s == 9 + len) {  // scattered eos (wins)
    float4 t = *(const float4*)(eos + b * H_ + e);
    v.x = t.x + padc.x; v.y = t.y + padc.y; v.z = t.z + padc.z; v.w = t.w + padc.w;
  } else if (s == 10 + len) {  // scattered bos (wins)
    v.x = padv.x + bosc.x; v.y = padv.y + bosc.y; v.z = padv.z + bosc.z; v.w = padv.w + bosc.w;
  } else if (s < 3) {  // role tokens
    v = *(const float4*)(embo + ((size_t)b * 131 + s) * H_ + e);
  } else if (s < 7) {  // THINK, TBOS, LANG, TEOS
    int j = s - 3;
    int cid = (j == 0) ? 8193 : (j == 1) ? 8194 : (j == 2) ? 8198 : 8195;
    float4 t = *(const float4*)(codec + (size_t)cid * H_ + e);
    v.x = padv.x + t.x; v.y = padv.y + t.y; v.z = padv.z + t.z; v.w = padv.w + t.w;
  } else if (s == 7) {  // speaker
    float4 t = *(const float4*)(spk + b * H_ + e);
    v.x = padv.x + t.x; v.y = padv.y + t.y; v.z = padv.z + t.z; v.w = padv.w + t.w;
  } else if (s == 8) {
    float4 t = *(const float4*)(bos + b * H_ + e);
    v.x = t.x + padc.x; v.y = t.y + padc.y; v.z = t.z + padc.z; v.w = t.w + padc.w;
  } else if (s < 9 + MAXT_) {  // text tokens + pad_e
    float4 t = *(const float4*)(embo + ((size_t)b * 131 + 3 + (s - 9)) * H_ + e);
    v.x = t.x + padc.x; v.y = t.y + padc.y; v.z = t.z + padc.z; v.w = t.w + padc.w;
  } else if (s == 9 + MAXT_) {
    float4 t = *(const float4*)(eos + b * H_ + e);
    v.x = t.x + padc.x; v.y = t.y + padc.y; v.z = t.z + padc.z; v.w = t.w + padc.w;
  } else {  // s == 138
    v.x = padv.x + bosc.x; v.y = padv.y + bosc.y; v.z = padv.z + bosc.z; v.w = padv.w + bosc.w;
  }
  *(float4*)(h + ((size_t)b * S_ + s) * H_ + e) = v;
}

// ============================================================
// 5) token RMSNorm (one block per token, one float4 per thread)
// ============================================================
__global__ __launch_bounds__(256) void k_rmsnorm(
    const float* __restrict__ x, const float* __restrict__ w,
    float* __restrict__ y) {
  int t = blockIdx.x;
  int e = threadIdx.x * 4;
  const float* xp = x + (size_t)t * H_;
  float4 v = *(const float4*)(xp + e);
  float ss = v.x * v.x + v.y * v.y + v.z * v.z + v.w * v.w;
#pragma unroll
  for (int off = 16; off; off >>= 1) ss += __shfl_xor(ss, off, 32);
  __shared__ float red[8];
  int wave = threadIdx.x >> 5, lane = threadIdx.x & 31;
  if (lane == 0) red[wave] = ss;
  __syncthreads();
  if (wave == 0) {
    float tot = (lane < 8) ? red[lane] : 0.f;
#pragma unroll
    for (int off = 4; off; off >>= 1) tot += __shfl_xor(tot, off, 32);
    if (lane == 0) red[0] = tot;
  }
  __syncthreads();
  float rms = rsqrtf(red[0] * (1.f / H_) + 1e-6f);
  float4 w4 = *(const float4*)(w + e);
  float4 o;
  o.x = v.x * rms * w4.x; o.y = v.y * rms * w4.y;
  o.z = v.z * rms * w4.z; o.w = v.w * rms * w4.w;
  *(float4*)(y + (size_t)t * H_ + e) = o;
}

// ============================================================
// 6) per-head RMSNorm + RoPE for q,k ; v transpose-copy to cache
//    grid = (24 head-slots, NTOK tokens), 64 threads
// ============================================================
__global__ __launch_bounds__(64) void k_qkv_rope(
    const float* __restrict__ qbuf, const float* __restrict__ kbuf,
    const float* __restrict__ vbuf, const float* __restrict__ qn,
    const float* __restrict__ kn, float* __restrict__ qr,
    float* __restrict__ kvK, float* __restrict__ kvV) {
  int tok = blockIdx.y;
  int b = tok / S_, s = tok % S_;
  int hslot = blockIdx.x;
  int d = threadIdx.x;
  if (hslot >= 20) {  // v: pure layout transform into kv cache
    int vh = hslot - 20;
    kvV[(((size_t)b * NKV_ + vh) * S_ + s) * HD_ + d] =
        vbuf[(size_t)tok * (NKV_ * HD_) + vh * HD_ + d];
    return;
  }
  bool isq = hslot < 16;
  float val = isq ? qbuf[(size_t)tok * (NH_ * HD_) + hslot * HD_ + d]
                  : kbuf[(size_t)tok * (NKV_ * HD_) + (hslot - 16) * HD_ + d];
  float ss = val * val;
#pragma unroll
  for (int off = 16; off; off >>= 1) ss += __shfl_xor(ss, off, 32);
  __shared__ float red2[2];
  __shared__ float sv[64];
  if ((threadIdx.x & 31) == 0) red2[threadIdx.x >> 5] = ss;
  __syncthreads();
  float tot = red2[0] + red2[1];
  float w = isq ? qn[d] : kn[d];
  val = val * rsqrtf(tot * (1.f / HD_) + 1e-6f) * w;
  sv[d] = val;
  __syncthreads();
  float partner = (d < 32) ? -sv[d + 32] : sv[d - 32];
  int j = d & 31;  // freq index (emb = concat(fr, fr))
  float freq = __expf(-(float)j * (13.815510557964274f / 32.f));  // theta=1e6
  float ang = (float)s * freq;
  float outv = val * __cosf(ang) + partner * __sinf(ang);
  if (isq) qr[(((size_t)b * NH_ + hslot) * S_ + s) * HD_ + d] = outv;
  else     kvK[(((size_t)b * NKV_ + (hslot - 16)) * S_ + s) * HD_ + d] = outv;
}

// ============================================================
// 7) WMMA attention: one block per (b, qh), 128 threads (4 waves)
// ============================================================
#define SPAD_ 160  // S padded to 5 K-chunks of 32
static constexpr size_t AOFF_SVT  = (size_t)4 * 16 * SPAD_ * 4;          // 40960
static constexpr size_t AOFF_SK   = AOFF_SVT + (size_t)64 * SPAD_ * 2;   // 61440
static constexpr size_t AOFF_SP   = AOFF_SK + (size_t)144 * 64 * 2;      // 79872
static constexpr size_t AOFF_SROW = AOFF_SP + (size_t)4 * 16 * SPAD_ * 2;// 100352
static constexpr size_t SMEM_ATTN = AOFF_SROW + 64 * 4;                  // 100608 B

__global__ __launch_bounds__(128) void k_attn(
    const float* __restrict__ qr, const float* __restrict__ kvK,
    const float* __restrict__ kvV, const int* __restrict__ tl,
    float* __restrict__ aout) {
  const int qh = blockIdx.x;
  const int b = blockIdx.y;
  const int kh = qh >> 2;  // GQA: 4 q heads per kv head
  const int tid = threadIdx.x, wave = tid >> 5, lane = tid & 31;
  const int hi = lane >> 4, ln = lane & 15;
  const int alen = tl[b] + 11;
  extern __shared__ char smem[];
  float*  sS   = (float*)smem;                  // per-wave 16 x SPAD f32
  __bf16* sVT  = (__bf16*)(smem + AOFF_SVT);    // [d][pos]  64 x SPAD
  __bf16* sK   = (__bf16*)(smem + AOFF_SK);     // [pos][d] 144 x 64
  __bf16* sP   = (__bf16*)(smem + AOFF_SP);     // per-wave 16 x SPAD bf16
  float*  sRow = (float*)(smem + AOFF_SROW);    // 64 x f32 (1/rowsum)
  const float* Kp = kvK + ((size_t)b * NKV_ + kh) * S_ * HD_;
  const float* Vp = kvV + ((size_t)b * NKV_ + kh) * S_ * HD_;
#pragma unroll
  for (int it = 0; it < 18; ++it) {  // K tile [pos][d], zero-padded rows
    int i = tid + it * 128;
    int pos = i >> 4, d4 = (i & 15) * 4;
    int cp = (pos < S_) ? pos : (S_ - 1);
    float4 v = *(const float4*)(Kp + (size_t)cp * HD_ + d4);
    if (pos >= S_) { v.x = 0.f; v.y = 0.f; v.z = 0.f; v.w = 0.f; }
    st_bf16x4(&sK[pos * 64 + d4], v);
  }
#pragma unroll
  for (int it = 0; it < 20; ++it) {  // V^T tile [d][pos], zero-padded cols
    int i = tid + it * 128;
    int pos = i >> 4, d4 = (i & 15) * 4;
    int cp = (pos < S_) ? pos : (S_ - 1);
    float4 v = *(const float4*)(Vp + (size_t)cp * HD_ + d4);
    if (pos >= S_) { v.x = 0.f; v.y = 0.f; v.z = 0.f; v.w = 0.f; }
    sVT[(d4 + 0) * SPAD_ + pos] = (__bf16)v.x;
    sVT[(d4 + 1) * SPAD_ + pos] = (__bf16)v.y;
    sVT[(d4 + 2) * SPAD_ + pos] = (__bf16)v.z;
    sVT[(d4 + 3) * SPAD_ + pos] = (__bf16)v.w;
  }
  __syncthreads();
  float* sSw = sS + wave * (16 * SPAD_);
  __bf16* sPw = sP + wave * (16 * SPAD_);
  for (int t = 0; t < 3; ++t) {  // 9 q-tiles over 4 waves
    int qt = wave + 4 * t;
    bool active = qt < 9;  // wave-uniform
    int q0 = qt * 16;
    if (active) {
      // ---- scores: Q(16x64) @ K^T, tiles of 16 kv positions ----
      int qm = q0 + ln; if (qm > S_ - 1) qm = S_ - 1;  // clamp pad rows
      const float* qp = qr + (((size_t)b * NH_ + qh) * S_ + qm) * HD_;
      bf16x16 aq0 = cvt_frag(qp + 16 * hi, qp + 8 + 16 * hi);
      bf16x16 aq1 = cvt_frag(qp + 32 + 16 * hi, qp + 40 + 16 * hi);
      for (int kt = 0; kt < 9; ++kt) {
        const __bf16* kp0 = &sK[(kt * 16 + ln) * 64 + 16 * hi];
        bf16x16 b0 = ld_b16x16(kp0);
        bf16x16 b1 = ld_b16x16(kp0 + 32);
        f32x8 acc = {};
        acc = wmma_bf16(aq0, b0, acc);
        acc = wmma_bf16(aq1, b1, acc);
#pragma unroll
        for (int j = 0; j < 8; ++j)
          sSw[(j + 8 * hi) * SPAD_ + kt * 16 + ln] = acc[j] * 0.125f;  // 1/sqrt(64)
      }
    }
    __syncthreads();
    if (active) {
      // ---- masked softmax per row (causal + pad), P stored bf16 ----
      for (int r = 0; r < 16; ++r) {
        int qpos = q0 + r;
        int lim = (qpos + 1 < alen) ? (qpos + 1) : alen;  // j valid iff j < lim
        float mx = -3.0e38f;
        for (int j = lane; j < SPAD_; j += 32)
          if (j < lim) mx = fmaxf(mx, sSw[r * SPAD_ + j]);
#pragma unroll
        for (int off = 16; off; off >>= 1) mx = fmaxf(mx, __shfl_xor(mx, off, 32));
        float sum = 0.f;
        for (int j = lane; j < SPAD_; j += 32) {
          float p = (j < lim) ? __expf(sSw[r * SPAD_ + j] - mx) : 0.f;
          sum += p;
          sPw[r * SPAD_ + j] = (__bf16)p;
        }
#pragma unroll
        for (int off = 16; off; off >>= 1) sum += __shfl_xor(sum, off, 32);
        if (lane == 0) sRow[wave * 16 + r] = 1.0f / sum;
      }
    }
    __syncthreads();
    if (active) {
      // ---- O = P(16xSPAD) @ V(SPADx64) ----
#pragma unroll
      for (int nt = 0; nt < 4; ++nt) {
        f32x8 acc = {};
#pragma unroll
        for (int kc = 0; kc < 5; ++kc) {
          const __bf16* pp = &sPw[ln * SPAD_ + kc * 32];
          bf16x16 ap = ld_b16x16(pp + 16 * hi);
          bf16x16 bv = ld_b16x16(&sVT[(nt * 16 + ln) * SPAD_ + kc * 32 + 16 * hi]);
          acc = wmma_bf16(ap, bv, acc);
        }
#pragma unroll
        for (int j = 0; j < 8; ++j) {
          int m = j + 8 * hi;
          int q = q0 + m;
          if (q < S_)
            aout[((size_t)b * S_ + q) * (NH_ * HD_) + qh * HD_ + nt * 16 + ln] =
                acc[j] * sRow[wave * 16 + m];
        }
      }
    }
    __syncthreads();
  }
}

// ============================================================
// 8) gather last hidden (post final norm)
// ============================================================
__global__ __launch_bounds__(256) void k_gather_last(
    const float* __restrict__ xn, const int* __restrict__ tl,
    float* __restrict__ lastb, float* __restrict__ out_last) {
  int b = blockIdx.x;
  int pos = tl[b] + 10;  // actual_len - 1
  int e = threadIdx.x * 4;
  float4 v = *(const float4*)(xn + ((size_t)b * S_ + pos) * H_ + e);
  *(float4*)(lastb + b * H_ + e) = v;
  *(float4*)(out_last + b * H_ + e) = v;
}

// ============================================================
// host orchestration
// ============================================================
extern "C" void kernel_launch(void* const* d_in, const int* in_sizes, int n_in,
                              void* d_out, int out_size, void* d_ws, size_t ws_size,
                              hipStream_t stream) {
  (void)n_in; (void)out_size; (void)ws_size;
  // ---- input binding: detect insertion-order vs pytree-sorted flattening ----
  const int *role_ids, *text_ids, *text_len;
  const float *tts_bos, *tts_pad, *tts_eos, *speaker;
  const float *text_emb, *text_proj, *codec_emb, *norm_w, *head_w;
  const float *q_w, *k_w, *v_w, *o_w, *qn, *kn, *in_ln, *post_ln, *gate_w, *up_w, *down_w;
  if (in_sizes[0] == 24) {  // setup_inputs() insertion order
    role_ids  = (const int*)d_in[0];   text_ids = (const int*)d_in[1];
    text_len  = (const int*)d_in[2];
    tts_bos   = (const float*)d_in[3]; tts_pad  = (const float*)d_in[4];
    tts_eos   = (const float*)d_in[5]; speaker  = (const float*)d_in[6];
    text_emb  = (const float*)d_in[7]; text_proj= (const float*)d_in[8];
    codec_emb = (const float*)d_in[9]; norm_w   = (const float*)d_in[10];
    head_w    = (const float*)d_in[11];
    q_w = (const float*)d_in[12]; k_w = (const float*)d_in[13];
    v_w = (const float*)d_in[14]; o_w = (const float*)d_in[15];
    qn  = (const float*)d_in[16]; kn  = (const float*)d_in[17];
    in_ln = (const float*)d_in[18]; post_ln = (const float*)d_in[19];
    gate_w = (const float*)d_in[20]; up_w = (const float*)d_in[21];
    down_w = (const float*)d_in[22];
  } else {  // jax pytree sorted-key order
    codec_emb = (const float*)d_in[0]; head_w  = (const float*)d_in[1];
    down_w = (const float*)d_in[2]; gate_w = (const float*)d_in[3];
    in_ln  = (const float*)d_in[4]; k_w    = (const float*)d_in[5];
    kn     = (const float*)d_in[6]; o_w    = (const float*)d_in[7];
    post_ln= (const float*)d_in[8]; q_w    = (const float*)d_in[9];
    qn     = (const float*)d_in[10]; up_w  = (const float*)d_in[11];
    v_w    = (const float*)d_in[12]; norm_w= (const float*)d_in[13];
    text_emb = (const float*)d_in[14]; text_proj = (const float*)d_in[15];
    role_ids = (const int*)d_in[16];   speaker   = (const float*)d_in[17];
    text_ids = (const int*)d_in[18];   text_len  = (const int*)d_in[19];
    tts_bos  = (const float*)d_in[20]; tts_eos   = (const float*)d_in[21];
    tts_pad  = (const float*)d_in[22];
  }
  // ---- output partitions (logits | kv_cache | last), all f32 ----
  float* out_logits = (float*)d_out;
  float* out_kv = out_logits + (size_t)B_ * CV_;
  float* out_last = out_kv + (size_t)2 * L_ * B_ * NKV_ * S_ * HD_;
  // ---- workspace carve (floats); GEMM A-operands padded to 64-row tiles ----
  float* ws = (float*)d_ws;
  float* embg = ws;                                  // NEMBP x 2048 (padded)
  float* embo = embg + (size_t)NEMBP_ * TD_;         // NEMB x 1024
  float* h    = embo + (size_t)NEMB_ * H_;           // B,S,H
  float* xn   = h    + (size_t)NTOK_ * H_;           // NTOKP x 1024 (padded)
  float* qb   = xn   + (size_t)NTOKP_ * H_;          // B,S,1024
  float* kb   = qb   + (size_t)NTOK_ * H_;           // B,S,256
  float* vb   = kb   + (size_t)NTOK_ * NKV_ * HD_;
  float* qrb  = vb   + (size_t)NTOK_ * NKV_ * HD_;   // B,NH,S,HD
  float* ao   = qrb  + (size_t)NTOK_ * H_;           // NTOKP x 1024 (padded)
  float* act  = ao   + (size_t)NTOKP_ * H_;          // NTOKP x 4096 (padded)
  float* lastb= act  + (size_t)NTOKP_ * FF_;         // 64 x 1024 (padded)

  const int MG = NTOKP_ / 64;   // 18 M-tiles for token GEMMs
  const int MGE = NEMBP_ / 64;  // 17 M-tiles for embedding GEMM

  // ---- embeddings ----
  k_gather_rows<<<NEMB_, 256, 0, stream>>>(text_emb, role_ids, text_ids, embg);
  k_gemm<<<dim3(H_ / 64, MGE), 128, 0, stream>>>(embg, text_proj, nullptr, embo,
                                                 NEMB_, H_, TD_);
  k_assemble<<<dim3(S_, B_), 256, 0, stream>>>(embo, codec_emb, tts_bos, tts_pad,
                                               tts_eos, speaker, text_len, h);
  // ---- transformer layers ----
  for (int l = 0; l < L_; ++l) {
    const float* qw_l = q_w + (size_t)l * H_ * NH_ * HD_;
    const float* kw_l = k_w + (size_t)l * H_ * NKV_ * HD_;
    const float* vw_l = v_w + (size_t)l * H_ * NKV_ * HD_;
    const float* ow_l = o_w + (size_t)l * NH_ * HD_ * H_;
    const float* gw_l = gate_w + (size_t)l * H_ * FF_;
    const float* uw_l = up_w + (size_t)l * H_ * FF_;
    const float* dw_l = down_w + (size_t)l * FF_ * H_;
    float* kvK_l = out_kv + (size_t)(2 * l) * B_ * NKV_ * S_ * HD_;
    float* kvV_l = out_kv + (size_t)(2 * l + 1) * B_ * NKV_ * S_ * HD_;

    k_rmsnorm<<<NTOK_, 256, 0, stream>>>(h, in_ln + (size_t)l * H_, xn);
    k_gemm<<<dim3(16, MG), 128, 0, stream>>>(xn, qw_l, nullptr, qb, NTOK_, NH_ * HD_, H_);
    k_gemm<<<dim3(4, MG), 128, 0, stream>>>(xn, kw_l, nullptr, kb, NTOK_, NKV_ * HD_, H_);
    k_gemm<<<dim3(4, MG), 128, 0, stream>>>(xn, vw_l, nullptr, vb, NTOK_, NKV_ * HD_, H_);
    k_qkv_rope<<<dim3(24, NTOK_), 64, 0, stream>>>(
        qb, kb, vb, qn + (size_t)l * HD_, kn + (size_t)l * HD_, qrb, kvK_l, kvV_l);
    k_attn<<<dim3(NH_, B_), 128, SMEM_ATTN, stream>>>(qrb, kvK_l, kvV_l, text_len, ao);
    k_gemm<<<dim3(16, MG), 128, 0, stream>>>(ao, ow_l, h, h, NTOK_, H_, NH_ * HD_);
    k_rmsnorm<<<NTOK_, 256, 0, stream>>>(h, post_ln + (size_t)l * H_, xn);
    k_gemm_gateup<<<dim3(FF_ / 64, MG), 128, 0, stream>>>(xn, gw_l, uw_l, act,
                                                          NTOK_, FF_, H_);
    k_gemm<<<dim3(16, MG), 128, 0, stream>>>(act, dw_l, h, h, NTOK_, H_, FF_);
  }
  // ---- head ----
  k_rmsnorm<<<NTOK_, 256, 0, stream>>>(h, norm_w, xn);
  k_gather_last<<<B_, 256, 0, stream>>>(xn, text_len, lastb, out_last);
  k_gemm<<<dim3(CV_ / 64, 1), 128, 0, stream>>>(lastb, head_w, nullptr, out_logits,
                                                B_, CV_, H_);
}
